// MultiHeadAttention_43258910605519
// MI455X (gfx1250) — matrix operand order repaired and checked by reading, not compiled, not run on hardware
//
#include <hip/hip_runtime.h>

// ---------------------------------------------------------------------------
// MHA forward for MI455X (gfx1250, wave32, WMMA).
// B=2, S=2048, D=1024, H=16, DK=64.  All matrix math on v_wmma_f32_16x16x32_bf16.
// ---------------------------------------------------------------------------

#define B_    2
#define S_    2048
#define D_    1024
#define H_    16
#define DK_   64
#define NROW_ 4096   // B_*S_

// Padded LDS row lengths (in bf16) — 16B-multiple strides whose dword stride
// (20 / 36) is conflict-free across 16 consecutive lanes (gcd with 64 banks = 4).
#define LDK32 40   // tiles with 32 bf16 of payload per row
#define LDK64 72   // tiles with 64 bf16 of payload per row

typedef __bf16 bf16;
typedef __attribute__((ext_vector_type(16))) __bf16   v16bf;
typedef __attribute__((ext_vector_type(8)))  float    v8f;
typedef __attribute__((ext_vector_type(4)))  unsigned u32x4;
typedef __attribute__((ext_vector_type(2)))  unsigned u32x2;

static_assert(sizeof(v16bf) == 32, "v16bf must be 32 bytes");

union FragU { v16bf v; u32x4 q[2]; };

// A-matrix fragment (16x32 bf16, MxK). Lane l holds row M=l&15.
// dwords: [kkdw+4*hi .. +3] (K 0..7 / 8..15), [kkdw+8+4*hi .. +3] (K 16..23 / 24..31)
__device__ __forceinline__ v16bf frag_a(const bf16* row, int hi, int kkdw) {
  const u32x4* p = reinterpret_cast<const u32x4*>(row);
  FragU f;
  f.q[0] = p[(kkdw + 4 * hi) >> 2];
  f.q[1] = p[(kkdw + 8 + 4 * hi) >> 2];
  return f.v;
}

// B-matrix fragment (32x16 bf16, KxN) from a Bt[n][k] (N-major, K-contig) row.
// Lane l holds col N=l&15; same K order as frag_a (K 0..7 / 8..15, then K 16..23 / 24..31).
__device__ __forceinline__ v16bf frag_b(const bf16* row, int hi, int kkdw) {
  const u32x4* p = reinterpret_cast<const u32x4*>(row);
  FragU f;
  f.q[0] = p[(kkdw + 4 * hi) >> 2];
  f.q[1] = p[(kkdw + 8 + 4 * hi) >> 2];
  return f.v;
}

__device__ __forceinline__ v8f wmma_bf16(v16bf a, v16bf b, v8f c) {
  return __builtin_amdgcn_wmma_f32_16x16x32_bf16(false, a, false, b, (short)0, c,
                                                 false, false);
}

// two fp32 -> one packed-bf16 dword (compiler emits v_cvt_pk_bf16_f32)
__device__ __forceinline__ unsigned pk2(float x, float y) {
  union { bf16 h[2]; unsigned u; } t;
  t.h[0] = (bf16)x;
  t.h[1] = (bf16)y;
  return t.u;
}

// ---------------------------------------------------------------------------
// Kernel 1: fused QKV projection + bias + RoPE, fp32 in, bf16 (B,H,S,DK) out.
// grid = (NROW_/64, D_/64, 3)  block = 128  (4 waves; wave w owns rows w*16..+15)
// ---------------------------------------------------------------------------
__global__ __launch_bounds__(128) void qkv_rope_kernel(
    const float* __restrict__ qin, const float* __restrict__ kin,
    const float* __restrict__ vin,
    const float* __restrict__ Wq, const float* __restrict__ bq,
    const float* __restrict__ Wk, const float* __restrict__ bk,
    const float* __restrict__ Wv, const float* __restrict__ bv,
    bf16* __restrict__ qh, bf16* __restrict__ kh, bf16* __restrict__ vh) {
  const int which = blockIdx.z;
  const float* X    = (which == 0) ? qin : (which == 1) ? kin : vin;
  const float* W    = (which == 0) ? Wq  : (which == 1) ? Wk  : Wv;
  const float* bias = (which == 0) ? bq  : (which == 1) ? bk  : bv;
  bf16* out         = (which == 0) ? qh  : (which == 1) ? kh  : vh;
  const bool do_rope = (which < 2);

  __shared__ __align__(16) bf16 As[64][LDK32];  // (M, K) payload 32
  __shared__ __align__(16) bf16 Bs[64][LDK32];  // (N, K) payload 32 (= rows of W)

  const int tid  = threadIdx.x;
  const int wave = tid >> 5;
  const int lane = tid & 31;
  const int hi   = lane >> 4;
  const int ln   = lane & 15;

  const int row0 = blockIdx.x * 64;  // token rows
  const int col0 = blockIdx.y * 64;  // output features (== head * 64)

  v8f acc[4] = {};

  for (int k0 = 0; k0 < D_; k0 += 32) {
    // --- phase 1: batched global b128 loads (overlap previous compute) ---
    float4 ta[4], tb[4];
#pragma unroll
    for (int p = 0; p < 4; ++p) {
      int i4 = tid + p * 128;          // 512 float4 per 64x32 tile
      int m = i4 >> 3, kq = (i4 & 7) << 2;
      ta[p] = *reinterpret_cast<const float4*>(&X[(size_t)(row0 + m) * D_ + k0 + kq]);
      tb[p] = *reinterpret_cast<const float4*>(&W[(size_t)(col0 + m) * D_ + k0 + kq]);
    }
    if (k0 + 32 < D_) {
      __builtin_prefetch(&X[(size_t)(row0 + (tid >> 1)) * D_ + k0 + 32], 0, 0);
      __builtin_prefetch(&W[(size_t)(col0 + (tid >> 1)) * D_ + k0 + 32], 0, 0);
    }
    __syncthreads();  // previous iteration's LDS reads complete
    // --- phase 2: pack fp32->bf16 and store b64 to LDS ---
#pragma unroll
    for (int p = 0; p < 4; ++p) {
      int i4 = tid + p * 128;
      int m = i4 >> 3, kq = (i4 & 7) << 2;
      u32x2 da; da.x = pk2(ta[p].x, ta[p].y); da.y = pk2(ta[p].z, ta[p].w);
      *reinterpret_cast<u32x2*>(&As[m][kq]) = da;
      u32x2 db; db.x = pk2(tb[p].x, tb[p].y); db.y = pk2(tb[p].z, tb[p].w);
      *reinterpret_cast<u32x2*>(&Bs[m][kq]) = db;
    }
    __syncthreads();
    // --- phase 3: batched fragment loads, then WMMA chain ---
    v16bf a = frag_a(&As[wave * 16 + ln][0], hi, 0);
    v16bf bfr[4];
#pragma unroll
    for (int nb = 0; nb < 4; ++nb) bfr[nb] = frag_b(&Bs[nb * 16 + ln][0], hi, 0);
#pragma unroll
    for (int nb = 0; nb < 4; ++nb) acc[nb] = wmma_bf16(a, bfr[nb], acc[nb]);
  }

  // bias
#pragma unroll
  for (int nb = 0; nb < 4; ++nb) {
    float bv_ = bias[col0 + nb * 16 + ln];
#pragma unroll
    for (int r = 0; r < 8; ++r) acc[nb][r] += bv_;
  }

  // RoPE (head dim 64 == this block's column extent; both halves in-register)
  if (do_rope) {
#pragma unroll
    for (int nb = 0; nb < 2; ++nb) {
      int j = nb * 16 + ln;  // 0..31
      float invf = __powf(10000.0f, -(float)j * (1.0f / 32.0f));
#pragma unroll
      for (int r = 0; r < 8; ++r) {
        int grow = row0 + wave * 16 + r + 8 * hi;
        int s = grow & (S_ - 1);
        float th = (float)s * invf, sn, cs;
        __sincosf(th, &sn, &cs);
        float x1 = acc[nb][r], x2 = acc[nb + 2][r];
        acc[nb][r]     = x1 * cs - x2 * sn;
        acc[nb + 2][r] = x1 * sn + x2 * cs;
      }
    }
  }

  // store bf16 to (B, H, S, DK)
  const int h = blockIdx.y;
#pragma unroll
  for (int nb = 0; nb < 4; ++nb) {
    int dk = nb * 16 + ln;
#pragma unroll
    for (int r = 0; r < 8; ++r) {
      int grow = row0 + wave * 16 + r + 8 * hi;
      int b = grow >> 11, s = grow & (S_ - 1);
      out[(((size_t)(b * H_ + h) * S_ + s) * DK_) + dk] = (bf16)acc[nb][r];
    }
  }
}

// ---------------------------------------------------------------------------
// Kernel 2: causal flash attention, bf16 in (B,H,S,DK), bf16 out (B,S,H*DK).
// grid = (S_/64, H_, B_)  block = 128.  Wave w owns q rows w*16..+15.
// ---------------------------------------------------------------------------
__global__ __launch_bounds__(128) void flash_attn_kernel(
    const bf16* __restrict__ qh, const bf16* __restrict__ kh,
    const bf16* __restrict__ vh, bf16* __restrict__ attn) {
  const int qb = blockIdx.x, h = blockIdx.y, b = blockIdx.z;

  __shared__ __align__(16) bf16 Qs[64][LDK64];     // (q-row, feat)
  __shared__ __align__(16) bf16 Ks[32][LDK64];     // (k-token, feat)  -> Bt scores
  __shared__ __align__(16) bf16 Vs[64][LDK32];     // (dk, k-token)    -> Bt for P*V
  __shared__ __align__(16) bf16 Ps[4][16][LDK32];  // per-wave P scratch (C->A)

  const int tid = threadIdx.x, wave = tid >> 5, lane = tid & 31;
  const int hi = lane >> 4, ln = lane & 15;
  const size_t head_base = (size_t)(b * H_ + h) * S_ * DK_;
  const int q0 = qb * 64;

  // Stage the Q tile once (64 rows x 64 feats = 512 u32x4; 4 per thread).
  {
    const u32x4* src =
        reinterpret_cast<const u32x4*>(qh + head_base + (size_t)q0 * DK_);
    u32x4 t[4];
#pragma unroll
    for (int p = 0; p < 4; ++p) t[p] = src[tid + p * 128];
#pragma unroll
    for (int p = 0; p < 4; ++p) {
      int i4 = tid + p * 128;
      int qr = i4 >> 2, dq = i4 & 3;          // 4 u32x4 (32 bf16) per Qs row
      reinterpret_cast<u32x4*>(&Qs[qr][0])[dq] = t[p];
    }
  }
  __syncthreads();

  v16bf qa[2];
  qa[0] = frag_a(&Qs[wave * 16 + ln][0], hi, 0);   // feat 0..31
  qa[1] = frag_a(&Qs[wave * 16 + ln][0], hi, 16);  // feat 32..63

  v8f oacc[4] = {};
  float m_run[8], l_run[8];
#pragma unroll
  for (int r = 0; r < 8; ++r) { m_run[r] = -1e30f; l_run[r] = 0.0f; }

  const int my_max_row = q0 + wave * 16 + 15;
  const int nkb = (qb + 1) * 2;  // 32-token k-blocks for causal coverage

  for (int jb = 0; jb < nkb; ++jb) {
    const int t0 = jb * 32;
    // --- batched global loads of K and V blocks (overlap previous compute) ---
    const u32x4* ksrc =
        reinterpret_cast<const u32x4*>(kh + head_base + (size_t)t0 * DK_);
    const u32x4* vsrc =
        reinterpret_cast<const u32x4*>(vh + head_base + (size_t)t0 * DK_);
    u32x4 kt[2], vt[2];
#pragma unroll
    for (int p = 0; p < 2; ++p) {
      kt[p] = ksrc[tid + p * 128];
      vt[p] = vsrc[tid + p * 128];
    }
    __syncthreads();  // previous block's LDS reads complete
    // --- stores: K direct (row-padded), V transposed to (dk, token) ---
#pragma unroll
    for (int p = 0; p < 2; ++p) {
      int i4 = tid + p * 128;
      int tok = i4 >> 2, dq = i4 & 3;
      reinterpret_cast<u32x4*>(&Ks[tok][0])[dq] = kt[p];
    }
#pragma unroll
    for (int p = 0; p < 2; ++p) {
      int i4 = tid + p * 128;
      int tok = i4 >> 3, dk0 = (i4 & 7) << 3;  // one u32x4 = 8 consecutive dk
      union { u32x4 q; bf16 hh[8]; } c;
      c.q = vt[p];
#pragma unroll
      for (int e = 0; e < 8; ++e) Vs[dk0 + e][tok] = c.hh[e];
    }
    __syncthreads();

    if (t0 <= my_max_row) {  // wave-uniform: EXEC stays all-ones for WMMA
      // ---- scores S = Q K^T (two 16x16 tiles over 32 tokens) ----
      v16bf kb0 = frag_b(&Ks[ln][0], hi, 0);
      v16bf kb1 = frag_b(&Ks[16 + ln][0], hi, 0);
      v16bf kb2 = frag_b(&Ks[ln][0], hi, 16);
      v16bf kb3 = frag_b(&Ks[16 + ln][0], hi, 16);
      v8f s0 = {}, s1 = {};
      s0 = wmma_bf16(qa[0], kb0, s0);
      s1 = wmma_bf16(qa[0], kb1, s1);
      s0 = wmma_bf16(qa[1], kb2, s0);
      s1 = wmma_bf16(qa[1], kb3, s1);

      // ---- scale + causal mask + online softmax ----
      float pm[8], psum[8], corr[8];
#pragma unroll
      for (int r = 0; r < 8; ++r) {
        int row = q0 + wave * 16 + r + 8 * hi;
        float v0 = s0[r] * 0.125f;  // 1/sqrt(64)
        float v1 = s1[r] * 0.125f;
        if (t0 + ln > row) v0 = -1e30f;
        if (t0 + 16 + ln > row) v1 = -1e30f;
        s0[r] = v0; s1[r] = v1;
        pm[r] = fmaxf(v0, v1);
      }
#pragma unroll
      for (int xm = 1; xm < 16; xm <<= 1)
#pragma unroll
        for (int r = 0; r < 8; ++r) pm[r] = fmaxf(pm[r], __shfl_xor(pm[r], xm, 32));
#pragma unroll
      for (int r = 0; r < 8; ++r) {
        float mn = fmaxf(m_run[r], pm[r]);
        corr[r] = __expf(m_run[r] - mn);
        m_run[r] = mn;
        float p0 = __expf(s0[r] - mn), p1 = __expf(s1[r] - mn);
        s0[r] = p0; s1[r] = p1;
        psum[r] = p0 + p1;
      }
#pragma unroll
      for (int xm = 1; xm < 16; xm <<= 1)
#pragma unroll
        for (int r = 0; r < 8; ++r) psum[r] += __shfl_xor(psum[r], xm, 32);
#pragma unroll
      for (int r = 0; r < 8; ++r) l_run[r] = l_run[r] * corr[r] + psum[r];
#pragma unroll
      for (int nb = 0; nb < 4; ++nb)
#pragma unroll
        for (int r = 0; r < 8; ++r) oacc[nb][r] *= corr[r];

      // ---- P (C-layout) -> LDS -> A-layout fragment, then O += P V ----
#pragma unroll
      for (int r = 0; r < 8; ++r) {
        Ps[wave][r + 8 * hi][ln]      = (bf16)s0[r];
        Ps[wave][r + 8 * hi][16 + ln] = (bf16)s1[r];
      }
      __builtin_amdgcn_wave_barrier();
      asm volatile("s_wait_dscnt 0" ::: "memory");  // intra-wave LDS RAW fence
      v16bf pa = frag_a(&Ps[wave][ln][0], hi, 0);
      v16bf vb[4];
#pragma unroll
      for (int nb = 0; nb < 4; ++nb) vb[nb] = frag_b(&Vs[nb * 16 + ln][0], hi, 0);
#pragma unroll
      for (int nb = 0; nb < 4; ++nb) oacc[nb] = wmma_bf16(pa, vb[nb], oacc[nb]);
    }
  }

  // Finalize: O /= l, store bf16 to flat (B*S, D) at (b, s, h, dk).
  float inv_l[8];
#pragma unroll
  for (int r = 0; r < 8; ++r) inv_l[r] = 1.0f / l_run[r];
#pragma unroll
  for (int nb = 0; nb < 4; ++nb) {
    int dk = nb * 16 + ln;
#pragma unroll
    for (int r = 0; r < 8; ++r) {
      int srow = q0 + wave * 16 + r + 8 * hi;
      attn[(size_t)(b * S_ + srow) * D_ + h * DK_ + dk] =
          (bf16)(oacc[nb][r] * inv_l[r]);
    }
  }
}

// ---------------------------------------------------------------------------
// Kernel 3: output projection, bf16 A x bf16(Wo^T) -> fp32 + bias.
// grid = (NROW_/64, D_/64)  block = 128.
// ---------------------------------------------------------------------------
__global__ __launch_bounds__(128) void out_proj_kernel(
    const bf16* __restrict__ Ain, const float* __restrict__ Wo,
    const float* __restrict__ bo, float* __restrict__ out) {
  __shared__ __align__(16) bf16 As[64][LDK32];
  __shared__ __align__(16) bf16 Bs[64][LDK32];

  const int tid = threadIdx.x, wave = tid >> 5, lane = tid & 31;
  const int hi = lane >> 4, ln = lane & 15;
  const int row0 = blockIdx.x * 64, col0 = blockIdx.y * 64;

  v8f acc[4] = {};

  for (int k0 = 0; k0 < D_; k0 += 32) {
    // batched loads: A (bf16, 256 u32x4 -> 2/thread), Wo (fp32, 4 float4/thread)
    u32x4 taq[2];
    float4 tb[4];
#pragma unroll
    for (int p = 0; p < 2; ++p) {
      int i4 = tid + p * 128;
      int m = i4 >> 2, dq = i4 & 3;  // 4 u32x4 per 32-bf16 row
      taq[p] = reinterpret_cast<const u32x4*>(
          Ain)[(((size_t)(row0 + m)) * D_ + k0) / 8 + dq];
    }
#pragma unroll
    for (int p = 0; p < 4; ++p) {
      int i4 = tid + p * 128;
      int m = i4 >> 3, kq = (i4 & 7) << 2;
      tb[p] = *reinterpret_cast<const float4*>(&Wo[(size_t)(col0 + m) * D_ + k0 + kq]);
    }
    __syncthreads();
#pragma unroll
    for (int p = 0; p < 2; ++p) {
      int i4 = tid + p * 128;
      int m = i4 >> 2, dq = i4 & 3;
      reinterpret_cast<u32x4*>(&As[m][0])[dq] = taq[p];
    }
#pragma unroll
    for (int p = 0; p < 4; ++p) {
      int i4 = tid + p * 128;
      int m = i4 >> 3, kq = (i4 & 7) << 2;
      u32x2 db; db.x = pk2(tb[p].x, tb[p].y); db.y = pk2(tb[p].z, tb[p].w);
      *reinterpret_cast<u32x2*>(&Bs[m][kq]) = db;
    }
    __syncthreads();

    v16bf a = frag_a(&As[wave * 16 + ln][0], hi, 0);
    v16bf bfr[4];
#pragma unroll
    for (int nb = 0; nb < 4; ++nb) bfr[nb] = frag_b(&Bs[nb * 16 + ln][0], hi, 0);
#pragma unroll
    for (int nb = 0; nb < 4; ++nb) acc[nb] = wmma_bf16(a, bfr[nb], acc[nb]);
  }

#pragma unroll
  for (int nb = 0; nb < 4; ++nb) {
    int col = col0 + nb * 16 + ln;
    float bias = bo[col];
#pragma unroll
    for (int r = 0; r < 8; ++r) {
      int grow = row0 + wave * 16 + r + 8 * hi;
      out[(size_t)grow * D_ + col] = acc[nb][r] + bias;
    }
  }
}

// ---------------------------------------------------------------------------
extern "C" void kernel_launch(void* const* d_in, const int* in_sizes, int n_in,
                              void* d_out, int out_size, void* d_ws, size_t ws_size,
                              hipStream_t stream) {
  (void)in_sizes; (void)n_in; (void)out_size; (void)ws_size;
  const float* q  = (const float*)d_in[0];
  const float* k  = (const float*)d_in[1];
  const float* v  = (const float*)d_in[2];
  // d_in[3] = mask: unused — causality is applied analytically in-kernel.
  const float* Wq = (const float*)d_in[4];
  const float* bq = (const float*)d_in[5];
  const float* Wk = (const float*)d_in[6];
  const float* bk = (const float*)d_in[7];
  const float* Wv = (const float*)d_in[8];
  const float* bv = (const float*)d_in[9];
  const float* Wo = (const float*)d_in[10];
  const float* bo = (const float*)d_in[11];

  char* ws = (char*)d_ws;
  bf16* qh   = (bf16*)(ws);                             //  8 MiB (B,H,S,DK)
  bf16* kh   = (bf16*)(ws + (size_t)8  * 1024 * 1024);
  bf16* vh   = (bf16*)(ws + (size_t)16 * 1024 * 1024);
  bf16* attn = (bf16*)(ws + (size_t)24 * 1024 * 1024);  //  8 MiB (B*S, D)

  dim3 blk(128);
  qkv_rope_kernel<<<dim3(NROW_ / 64, D_ / 64, 3), blk, 0, stream>>>(
      q, k, v, Wq, bq, Wk, bk, Wv, bv, qh, kh, vh);
  flash_attn_kernel<<<dim3(S_ / 64, H_, B_), blk, 0, stream>>>(qh, kh, vh, attn);
  out_proj_kernel<<<dim3(NROW_ / 64, D_ / 64), blk, 0, stream>>>(
      attn, Wo, bo, (float*)d_out);
}
